// LinearAttention_68882685493758
// MI455X (gfx1250) — matrix operand order repaired and checked by reading, not compiled, not run on hardware
//
#include <hip/hip_runtime.h>
#include <hip/hip_bf16.h>
#include <math.h>

typedef __attribute__((ext_vector_type(16))) __bf16 v16bf;
typedef __attribute__((ext_vector_type(8)))  __bf16 v8bf;
typedef __attribute__((ext_vector_type(2)))  __bf16 v2bf;
typedef __attribute__((ext_vector_type(8)))  float  v8f;

#define N_ 4
#define L_ 8192
#define S_ 8192
#define H_ 8
#define D_ 64
#define M_ 64
#define EPS_ 1e-6f

#define SCHUNK 512   // S rows per workgroup (phase 1)
#define STILE  128   // S rows per LDS tile
#define LTILE  128   // L rows per workgroup (phase 2)

__device__ __forceinline__ float feat(float x) {
    // elu(x)+1 : x>0 ? x+1 : exp(x)
    return x > 0.0f ? x + 1.0f : __expf(x);
}

// Concatenate two contiguous 16B LDS runs into one 16-element bf16 fragment.
__device__ __forceinline__ v16bf frag_cat(const __bf16* p0, const __bf16* p1) {
    v8bf lo = *(const v8bf*)p0;   // ds_load_b128
    v8bf hi = *(const v8bf*)p1;   // ds_load_b128
    return __builtin_shufflevector(lo, hi, 0, 1, 2, 3, 4, 5, 6, 7,
                                            8, 9, 10, 11, 12, 13, 14, 15);
}

__global__ void la_zero_ws(float* __restrict__ ws, int n) {
    int i = blockIdx.x * blockDim.x + threadIdx.x;
    if (i < n) ws[i] = 0.0f;
}

// Phase 1: KV[nh][m][d] += sum_s V[s,m]*K[s,d] ; Ksum[nh][d] += sum_s K[s,d]
__global__ void __launch_bounds__(256)
la_kv_kernel(const float* __restrict__ keys,    // [N,S,H,D]
             const float* __restrict__ values,  // [N,S,H,M]
             const float* __restrict__ mask,    // [N,S]
             float* __restrict__ kv_ws,         // [N*H][M][D]
             float* __restrict__ ksum_ws)       // [N*H][D]
{
    // Transposed staging: fragment-major so frag loads are ds_load_b128.
    __shared__ __attribute__((aligned(16))) __bf16 Kt[D_][STILE]; // B source: [d][s]
    __shared__ __attribute__((aligned(16))) __bf16 Vt[M_][STILE]; // A source: [m][s]

    const int nchunks = S_ / SCHUNK;
    const int pair  = blockIdx.x / nchunks;
    const int chunk = blockIdx.x % nchunks;
    const int n = pair / H_;
    const int h = pair % H_;

    const int tid  = threadIdx.x;
    const int lane = tid & 31;
    const int wave = tid >> 5;
    const int half = lane >> 4;
    const int l16  = lane & 15;

    // Each wave owns two adjacent 16x16 tiles of the 64x64 KV matrix.
    const int mt  = wave >> 1;         // M-tile (rows of KV)
    const int dt0 = (wave & 1) * 2;    // first D-tile (cols of KV)

    v8f c0 = {}; v8f c1 = {};
    float ksum_acc = 0.0f;             // this thread's staged col is fixed: tid&63
    const int d_own = tid & 63;

    const int s_base = chunk * SCHUNK;

    for (int st = 0; st < SCHUNK; st += STILE) {
        // Stage K (feature map * mask, transposed) and V (transposed) as bf16.
        // Pack s-pairs -> b32 LDS stores; fold Ksum into the staging pass.
        for (int i = tid; i < (STILE / 2) * 64; i += 256) {
            int col = i & 63;              // == tid&63 every iteration
            int sl  = (i >> 6) * 2;        // even local s
            int sg  = s_base + st + sl;
            size_t r0 = ((size_t)(n * S_ + sg) * H_ + h);
            size_t r1 = ((size_t)(n * S_ + sg + 1) * H_ + h);
            float k0 = feat(keys[r0 * D_ + col]) * mask[n * S_ + sg];
            float k1 = feat(keys[r1 * D_ + col]) * mask[n * S_ + sg + 1];
            ksum_acc += k0 + k1;
            *(v2bf*)&Kt[col][sl] = (v2bf){(__bf16)k0, (__bf16)k1};
            float v0 = values[r0 * M_ + col];
            float v1 = values[r1 * M_ + col];
            *(v2bf*)&Vt[col][sl] = (v2bf){(__bf16)v0, (__bf16)v1};
        }
        __syncthreads();

        // 4 WMMA K-steps of 32 over the 128-row tile; 2 tiles per wave.
        const int arow = mt * 16 + l16;
        const int bcol0 = dt0 * 16 + l16;
        const int bcol1 = (dt0 + 1) * 16 + l16;
#pragma unroll
        for (int s0 = 0; s0 < STILE; s0 += 32) {
            v16bf a  = frag_cat(&Vt[arow][s0 + half * 16],
                                &Vt[arow][s0 + half * 16 + 8]);
            v16bf b0 = frag_cat(&Kt[bcol0][s0 + half * 16],
                                &Kt[bcol0][s0 + half * 16 + 8]);
            v16bf b1 = frag_cat(&Kt[bcol1][s0 + half * 16],
                                &Kt[bcol1][s0 + half * 16 + 8]);
            c0 = __builtin_amdgcn_wmma_f32_16x16x32_bf16(false, a, false, b0,
                                                         (short)0, c0, false, false);
            c1 = __builtin_amdgcn_wmma_f32_16x16x32_bf16(false, a, false, b1,
                                                         (short)0, c1, false, false);
        }
        __syncthreads();
    }

    // Commit partial results with native f32 global atomics.
    float* kvp = kv_ws + (size_t)pair * (M_ * D_);
#pragma unroll
    for (int r = 0; r < 8; ++r) {
        int mrow = mt * 16 + half * 8 + r;
        __hip_atomic_fetch_add(&kvp[mrow * D_ + dt0 * 16 + l16], c0[r],
                               __ATOMIC_RELAXED, __HIP_MEMORY_SCOPE_AGENT);
        __hip_atomic_fetch_add(&kvp[mrow * D_ + (dt0 + 1) * 16 + l16], c1[r],
                               __ATOMIC_RELAXED, __HIP_MEMORY_SCOPE_AGENT);
    }
    __hip_atomic_fetch_add(&ksum_ws[pair * D_ + d_own], ksum_acc,
                           __ATOMIC_RELAXED, __HIP_MEMORY_SCOPE_AGENT);
}

// Phase 2: out[l,m] = Z[l] * sum_d Q[l,d] * KV[m,d]
__global__ void __launch_bounds__(256)
la_out_kernel(const float* __restrict__ queries, // [N,L,H,D]
              const float* __restrict__ kv_ws,   // [N*H][M][D]
              const float* __restrict__ ksum_ws, // [N*H][D]
              float* __restrict__ out)           // [N,L,H,M]
{
    __shared__ __attribute__((aligned(16))) __bf16 KVl[M_][D_];   // B source: [m][d]
    __shared__ __attribute__((aligned(16))) __bf16 Ql[LTILE][D_]; // A source: [l][d]
    __shared__ float Ks[D_];
    __shared__ float Zl[LTILE];

    const int ltiles = L_ / LTILE;    // 64
    const int pair = blockIdx.x / ltiles;
    const int lt   = blockIdx.x % ltiles;
    const int n = pair / H_;
    const int h = pair % H_;
    const int l_base = lt * LTILE;

    const int tid  = threadIdx.x;
    const int lane = tid & 31;
    const int wave = tid >> 5;
    const int half = lane >> 4;
    const int l16  = lane & 15;

    // Ksum first (needed by the fused Z computation below).
    if (tid < D_) Ks[tid] = ksum_ws[pair * D_ + tid];

    // KV -> LDS bf16 [m][d] (this IS B-column-major for B = KV^T).
    for (int i = tid; i < (M_ * D_) / 2; i += 256) {
        int m  = i >> 5;
        int dp = (i & 31) * 2;
        const float* kvr = &kv_ws[(size_t)pair * (M_ * D_) + m * D_ + dp];
        *(v2bf*)&KVl[m][dp] = (v2bf){(__bf16)kvr[0], (__bf16)kvr[1]};
    }
    __syncthreads();

    // Q staging with fused Z: each wave stages one row per step (lane = d-pair),
    // computes the Q.Ksum partial in f32, wave-reduces, lane 0 writes Z.
#pragma unroll
    for (int j = 0; j < LTILE / 8; ++j) {
        int row = wave + 8 * j;
        int lg  = l_base + row;
        const float* qr = &queries[((size_t)(n * L_ + lg) * H_ + h) * D_];
        float q0 = feat(qr[2 * lane]);
        float q1 = feat(qr[2 * lane + 1]);
        *(v2bf*)&Ql[row][2 * lane] = (v2bf){(__bf16)q0, (__bf16)q1};
        float part = q0 * Ks[2 * lane] + q1 * Ks[2 * lane + 1];
#pragma unroll
        for (int mdist = 16; mdist >= 1; mdist >>= 1)
            part += __shfl_xor(part, mdist, 32);
        if (lane == 0) Zl[row] = 1.0f / (part + EPS_);
    }
    __syncthreads();

    // Each wave: 16 L-rows x 64 M-cols = 4 tiles, K=64 -> 2 WMMAs/tile.
    v8f c[4] = {{}, {}, {}, {}};
    const int lrow = wave * 16;
    const int arow = lrow + l16;

#pragma unroll
    for (int kk = 0; kk < 2; ++kk) {
        v16bf a = frag_cat(&Ql[arow][kk * 32 + half * 16],
                           &Ql[arow][kk * 32 + half * 16 + 8]);
#pragma unroll
        for (int mtile = 0; mtile < 4; ++mtile) {
            int bcol = mtile * 16 + l16;
            v16bf b = frag_cat(&KVl[bcol][kk * 32 + half * 16],
                               &KVl[bcol][kk * 32 + half * 16 + 8]);
            c[mtile] = __builtin_amdgcn_wmma_f32_16x16x32_bf16(false, a, false, b,
                                                               (short)0, c[mtile],
                                                               false, false);
        }
    }

    // Scale by Z and store (f32).
#pragma unroll
    for (int mtile = 0; mtile < 4; ++mtile) {
#pragma unroll
        for (int r = 0; r < 8; ++r) {
            int ll = lrow + half * 8 + r;
            int lg = l_base + ll;
            out[((size_t)(n * L_ + lg) * H_ + h) * M_ + mtile * 16 + l16] =
                c[mtile][r] * Zl[ll];
        }
    }
}

extern "C" void kernel_launch(void* const* d_in, const int* in_sizes, int n_in,
                              void* d_out, int out_size, void* d_ws, size_t ws_size,
                              hipStream_t stream) {
    const float* q    = (const float*)d_in[0];
    const float* k    = (const float*)d_in[1];
    const float* v    = (const float*)d_in[2];
    const float* mask = (const float*)d_in[3];
    float* out = (float*)d_out;

    float* kv_ws   = (float*)d_ws;                  // 32*64*64 floats
    float* ksum_ws = kv_ws + N_ * H_ * M_ * D_;     // 32*64 floats

    const int zn = N_ * H_ * (M_ * D_ + D_);
    la_zero_ws<<<(zn + 255) / 256, 256, 0, stream>>>((float*)d_ws, zn);

    const int nchunks = S_ / SCHUNK;
    la_kv_kernel<<<N_ * H_ * nchunks, 256, 0, stream>>>(k, v, mask, kv_ws, ksum_ws);

    const int ltiles = L_ / LTILE;
    la_out_kernel<<<N_ * H_ * ltiles, 256, 0, stream>>>(q, kv_ws, ksum_ws, out);
}
